// GroupedRotaryShiftedWindowMultiHeadAttention_2422361555256
// MI455X (gfx1250) — matrix operand order repaired and checked by reading, not compiled, not run on hardware
//
#include <hip/hip_runtime.h>
#include <hip/hip_bf16.h>

// ---------------------------------------------------------------------------
// GroupedRotaryShiftedWindowMultiHeadAttention for MI455X (gfx1250, wave32)
// All matmuls via v_wmma_f32_16x16x32_bf16 (bf16 in, f32 accumulate).
// Async global->LDS staging (ASYNCcnt) used for the bf16-A GEMM; the builtin
// signature is (int4 as1* src, int4 as3* lds, imm offset, imm cpol).
// ---------------------------------------------------------------------------

typedef __attribute__((ext_vector_type(16))) __bf16 v16bf;
typedef __attribute__((ext_vector_type(8)))  __bf16 v8bf;
typedef __attribute__((ext_vector_type(8)))  float  v8f;
typedef int v4i __attribute__((ext_vector_type(4)));
typedef __attribute__((address_space(1))) v4i* gv4p;  // global int4*
typedef __attribute__((address_space(3))) v4i* lv4p;  // LDS int4*

#define H_DIM 2048
#define NHEAD 16
#define NGRP  4
#define WSZ   256
#define BATCH 2
#define NSEQ  4096
#define CGRP  512   // H/NG
#define HD    32    // CGRP/NH
#define NWIN  16    // NSEQ/WSZ
#define QDIM  128   // H/NH
#define KDIM  32    // H/(NH*NG)

#if defined(__HIP_DEVICE_COMPILE__) && \
    __has_builtin(__builtin_amdgcn_global_load_async_to_lds_b128)
#define HAVE_ASYNC_LDS 1
#endif

static __device__ __forceinline__ void wait_asynccnt0() {
#if defined(__HIP_DEVICE_COMPILE__) && __has_builtin(__builtin_amdgcn_s_wait_asynccnt)
  __builtin_amdgcn_s_wait_asynccnt(0);
#else
  asm volatile("s_wait_asynccnt 0x0" ::: "memory");
#endif
}

static __device__ __forceinline__ v16bf pack16(v8bf lo, v8bf hi8) {
  v16bf r;
#pragma unroll
  for (int i = 0; i < 8; ++i) { r[i] = lo[i]; r[i + 8] = hi8[i]; }
  return r;
}

// A fragment (16x32 bf16), CDNA5 layout: lane m = L&15; k = (L>=16?8:0)+e (e<8)
// and 16+(L>=16?8:0)+(e-8) (e>=8). base = row-major tile, ld elements.
static __device__ __forceinline__ v16bf load_a_frag(const __bf16* base, int ld,
                                                    int ln, int hi) {
  const __bf16* r = base + (size_t)ln * ld;
  v8bf x0 = *(const v8bf*)(r + hi * 8);
  v8bf x1 = *(const v8bf*)(r + 16 + hi * 8);
  return pack16(x0, x1);
}

// B fragment (32x16 bf16): lane holds column n; k = hi*8 + e (e<8), 16 + hi*8 + (e-8)
// (e>=8). rowptr points at this lane's 32-element k-span (contiguous).
static __device__ __forceinline__ v16bf load_b_frag(const __bf16* rowptr, int hi) {
  v8bf x0 = *(const v8bf*)(rowptr + hi * 8);
  v8bf x1 = *(const v8bf*)(rowptr + 16 + hi * 8);
  return pack16(x0, x1);
}

static __device__ __forceinline__ v8f wmma_bf16(v16bf a, v16bf b, v8f c) {
  return __builtin_amdgcn_wmma_f32_16x16x32_bf16(false, a, false, b, (short)0, c,
                                                 false, false);
}

// ---------------------------------------------------------------------------
// Weight convert: W[k][n] (f32) -> Wt[n][k] (bf16), so B-frags are row reads.
// ---------------------------------------------------------------------------
__global__ void wconv_kernel(const float* __restrict__ src, __bf16* __restrict__ dst,
                             int K, int Nout) {
  int idx = blockIdx.x * 256 + threadIdx.x;
  if (idx >= K * Nout) return;
  int k = idx / Nout, n = idx - k * Nout;
  dst[(size_t)n * K + k] = (__bf16)src[idx];
}

// ---------------------------------------------------------------------------
// Generic WMMA GEMM: C[m][n] = sum_k A[m][k] * Wt[n][k] + bias[n]
// Block: 256 threads (8 waves); wave w -> n-tile blockIdx.y*128 + w*16.
// A staged in LDS per 16x32 k-slab: f32 A converted via VALU; bf16 A copied
// with GLOBAL_LOAD_ASYNC_TO_LDS_B128 (ASYNCcnt) when available.
// ---------------------------------------------------------------------------
template <bool A_F32>
__global__ __launch_bounds__(256) void gemm_kernel(const void* __restrict__ Aptr,
                                                   const __bf16* __restrict__ Wt,
                                                   const float* __restrict__ bias,
                                                   float* __restrict__ Cout,
                                                   int K, int Nout) {
  __shared__ __bf16 Atile[16 * 32];
  const int tid = threadIdx.x, wave = tid >> 5, lane = tid & 31;
  const int ln = lane & 15, hi = lane >> 4;
  const int m0 = blockIdx.x * 16;
  const int ntile = blockIdx.y * 128 + wave * 16;
  const bool active = (ntile < Nout);
  const int sr = tid >> 4, sc = (tid * 2) & 31;  // 2 elems/thread staging

  v8f acc = {};
  for (int k0 = 0; k0 < K; k0 += 32) {
    __syncthreads();
    if constexpr (A_F32) {
      const float* A = (const float*)Aptr;
      const float* p = A + (size_t)(m0 + sr) * K + k0 + sc;
      Atile[sr * 32 + sc]     = (__bf16)p[0];
      Atile[sr * 32 + sc + 1] = (__bf16)p[1];
    } else {
      const __bf16* A = (const __bf16*)Aptr;
#ifdef HAVE_ASYNC_LDS
      // 16x32 bf16 tile = 1KB: 64 lanes x 16B async DMA into LDS.
      if (tid < 64) {
        const int r = tid >> 2, ch = (tid & 3) * 8;
        const __bf16* g = A + (size_t)(m0 + r) * K + k0 + ch;
        __builtin_amdgcn_global_load_async_to_lds_b128(
            (gv4p)g, (lv4p)&Atile[r * 32 + ch], 0, 0);
      }
      wait_asynccnt0();
#else
      const __bf16* p = A + (size_t)(m0 + sr) * K + k0 + sc;
      Atile[sr * 32 + sc]     = p[0];
      Atile[sr * 32 + sc + 1] = p[1];
#endif
    }
    __syncthreads();
    if (active) {
      // prefetch next k-slab of this lane's weight row while we compute
      if (k0 + 32 < K)
        __builtin_prefetch(Wt + (size_t)(ntile + ln) * K + k0 + 32, 0, 1);
      v16bf a = load_a_frag(Atile, 32, ln, hi);
      v16bf b = load_b_frag(Wt + (size_t)(ntile + ln) * K + k0, hi);
      acc = wmma_bf16(a, b, acc);
    }
  }
  if (active) {
    const float bv = bias[ntile + ln];
#pragma unroll
    for (int v = 0; v < 8; ++v)
      Cout[(size_t)(m0 + v + 8 * hi) * Nout + ntile + ln] = acc[v] + bv;
  }
}

// ---------------------------------------------------------------------------
// RoPE + channel-repeat(16) + grouping: emits bf16 vectors laid out so that
// attention Q A-frags and K/V B-frags are contiguous loads.
//   qr[((g*B+b)*N+n)*512 + c] = rope(q[b,n, g*32 + c/16])
//   kr[((b)*N+n)*512 + c]     = rope(k[b,n, c/16])
//   vr[((b)*N+n)*512 + c]     =       v[b,n, c/16]
// ---------------------------------------------------------------------------
__global__ void rope_kernel(const float* __restrict__ qf, const float* __restrict__ kf,
                            const float* __restrict__ vf, __bf16* __restrict__ qr,
                            __bf16* __restrict__ kr, __bf16* __restrict__ vr) {
  const size_t QR = (size_t)NGRP * BATCH * NSEQ * CGRP;  // 16777216
  const size_t KR = (size_t)BATCH * NSEQ * CGRP;         // 4194304
  size_t idx = (size_t)blockIdx.x * 256 + threadIdx.x;
  if (idx < QR) {
    int c = idx & 511; int n = (int)(idx >> 9) & 4095;
    int b = (int)(idx >> 21) & 1; int g = (int)(idx >> 22);
    const float* row = qf + ((size_t)b * NSEQ + n) * QDIM + g * HD;
    float x = row[c >> 4];
    int cp = (c < 256) ? c + 256 : c - 256;
    float rot = row[cp >> 4];
    if (c < 256) rot = -rot;
    float ang = (float)n * powf(10000.0f, -(float)(c & 255) * (1.0f / 256.0f));
    float s, co; sincosf(ang, &s, &co);
    qr[idx] = (__bf16)(x * co + rot * s);
  } else if (idx < QR + KR) {
    size_t i2 = idx - QR;
    int c = (int)(i2 & 511); int n = (int)(i2 >> 9) & 4095; int b = (int)(i2 >> 21) & 1;
    const float* row = kf + ((size_t)b * NSEQ + n) * KDIM;
    float x = row[c >> 4];
    int cp = (c < 256) ? c + 256 : c - 256;
    float rot = row[cp >> 4];
    if (c < 256) rot = -rot;
    float ang = (float)n * powf(10000.0f, -(float)(c & 255) * (1.0f / 256.0f));
    float s, co; sincosf(ang, &s, &co);
    kr[i2] = (__bf16)(x * co + rot * s);
  } else if (idx < QR + 2 * KR) {
    size_t i2 = idx - QR - KR;
    int c = (int)(i2 & 511); int n = (int)(i2 >> 9) & 4095; int b = (int)(i2 >> 21) & 1;
    vr[i2] = (__bf16)vf[((size_t)b * NSEQ + n) * KDIM + (c >> 4)];
  }
}

// ---------------------------------------------------------------------------
// Attention: one block per (g, w, b, h) = 2048 blocks, 128 threads (4 waves).
// "Head" h under the faithful reshape means: Qh[i,d] = qr-vec(n0+16h + i/16)
// at offset (i%16)*32 + d. So each strip of 16 query rows is one roped vector
// viewed as a 16x32 A tile. Multiplicative tril mask; zeros join the softmax.
// ---------------------------------------------------------------------------
__global__ __launch_bounds__(128, 1) void attn_kernel(const __bf16* __restrict__ qr,
                                                      const __bf16* __restrict__ kr,
                                                      const __bf16* __restrict__ vr,
                                                      float* __restrict__ attn_out,
                                                      __bf16* __restrict__ cv) {
  const int bid = blockIdx.x;
  const int h = bid & 15, b = (bid >> 4) & 1, w = (bid >> 5) & 15, g = bid >> 9;
  const int n0 = w * WSZ;
  const int tid = threadIdx.x, wave = tid >> 5, lane = tid & 31;
  const int ln = lane & 15, hi = lane >> 4;

  __shared__ __bf16 VT[32][264];        // V^T (pad keeps rows 16B-aligned)
  __shared__ __bf16 Pbuf[4][16][256];   // per-wave probability strip (bf16)

  // Stage VT[d][j] = Vh[j][d]
  for (int idx = tid; idx < 32 * 256; idx += 128) {
    int d = idx >> 8, j = idx & 255;
    VT[d][j] = vr[((size_t)b * NSEQ + n0 + 16 * h + (j >> 4)) * CGRP + (j & 15) * HD + d];
  }
  __syncthreads();

  const float scale = 0.044194173824159216f;  // 1/sqrt(512)
  const size_t abase =
      ((((size_t)g * NWIN + w) * BATCH + b) * NHEAD + h) * (size_t)(WSZ * WSZ);

  for (int sidx = 0; sidx < 4; ++sidx) {
    const int strip = wave * 4 + sidx;  // 0..15, query rows [16*strip, 16*strip+16)

    // Q A-fragment: one roped vector reinterpreted as a 16x32 tile.
    const __bf16* qbase =
        qr + (((size_t)g * BATCH + b) * NSEQ + n0 + 16 * h + strip) * CGRP;
    v16bf aq = load_a_frag(qbase, 32, ln, hi);

    // S strip = Q(16x32) . K^T, one WMMA per 16-column tile (k-depth = HD = 32).
    v8f S[16];
#pragma unroll
    for (int t = 0; t < 16; ++t) {
      const __bf16* krow =
          kr + ((size_t)b * NSEQ + n0 + 16 * h + t) * CGRP + ln * HD;
      v16bf bkf = load_b_frag(krow, hi);
      v8f z = {};
      S[t] = wmma_bf16(aq, bkf, z);
    }

    // Softmax: lane owns rows r = v + 8*hi of the strip, columns j = t*16 + ln.
#pragma unroll
    for (int v = 0; v < 8; ++v) {
      const int i = strip * 16 + v + 8 * hi;  // query index in window
      float mx = -3.4e38f;
#pragma unroll
      for (int t = 0; t < 16; ++t) {
        const int j = t * 16 + ln;
        float sv = (j <= i) ? S[t][v] * scale : 0.0f;  // multiplicative mask
        S[t][v] = sv;
        mx = fmaxf(mx, sv);
      }
      mx = fmaxf(mx, __shfl_xor(mx, 1, 32));
      mx = fmaxf(mx, __shfl_xor(mx, 2, 32));
      mx = fmaxf(mx, __shfl_xor(mx, 4, 32));
      mx = fmaxf(mx, __shfl_xor(mx, 8, 32));
      float sum = 0.0f;
#pragma unroll
      for (int t = 0; t < 16; ++t) {
        float p = __expf(S[t][v] - mx);
        S[t][v] = p;
        sum += p;
      }
      sum += __shfl_xor(sum, 1, 32);
      sum += __shfl_xor(sum, 2, 32);
      sum += __shfl_xor(sum, 4, 32);
      sum += __shfl_xor(sum, 8, 32);
      const float inv = 1.0f / sum;
      float* arow = attn_out + abase + (size_t)i * WSZ + ln;
      const int lrow = v + 8 * hi;
#pragma unroll
      for (int t = 0; t < 16; ++t) {
        float a = S[t][v] * inv;
        arow[t * 16] = a;                        // attention map (f32)
        Pbuf[wave][lrow][t * 16 + ln] = (__bf16)a;
      }
    }
    // Same-wave LDS RAW across lanes: drain DS pipe before fragment reloads.
    asm volatile("s_wait_dscnt 0x0" ::: "memory");

    // O(16x32) = P(16x256) . V(256x32), 8 k-chunks x 2 output col-tiles.
    v8f O0 = {}, O1 = {};
#pragma unroll
    for (int c = 0; c < 8; ++c) {
      v16bf pa = load_a_frag(&Pbuf[wave][0][c * 32], 256, ln, hi);
      v16bf b0 = load_b_frag(&VT[ln][c * 32], hi);
      O0 = wmma_bf16(pa, b0, O0);
      v16bf b1 = load_b_frag(&VT[16 + ln][c * 32], hi);
      O1 = wmma_bf16(pa, b1, O1);
    }
#pragma unroll
    for (int v = 0; v < 8; ++v) {
      const int i = strip * 16 + v + 8 * hi;
      __bf16* crow = cv + ((size_t)b * NSEQ + n0 + i) * H_DIM + g * CGRP + h * HD;
      crow[ln]      = (__bf16)O0[v];
      crow[16 + ln] = (__bf16)O1[v];
    }
  }
}

// ---------------------------------------------------------------------------
extern "C" void kernel_launch(void* const* d_in, const int* in_sizes, int n_in,
                              void* d_out, int out_size, void* d_ws, size_t ws_size,
                              hipStream_t stream) {
  (void)in_sizes; (void)n_in; (void)out_size; (void)ws_size;
  const float* inputs  = (const float*)d_in[0];
  const float* context = (const float*)d_in[1];
  // d_in[2] = mask: accepted by the reference but never used in the math.
  const float* Wq = (const float*)d_in[3];
  const float* bq = (const float*)d_in[4];
  const float* Wk = (const float*)d_in[5];
  const float* bk = (const float*)d_in[6];
  const float* Wv = (const float*)d_in[7];
  const float* bv = (const float*)d_in[8];
  const float* Wo = (const float*)d_in[9];
  const float* bo = (const float*)d_in[10];

  char* ws = (char*)d_ws;
  size_t off = 0;
  auto carve = [&](size_t bytes) -> char* {
    char* p = ws + off;
    off += (bytes + 255) & ~(size_t)255;
    return p;
  };
  __bf16* Wqt = (__bf16*)carve((size_t)QDIM * H_DIM * 2);    // 0.5 MB
  __bf16* Wkt = (__bf16*)carve((size_t)KDIM * H_DIM * 2);
  __bf16* Wvt = (__bf16*)carve((size_t)KDIM * H_DIM * 2);
  __bf16* Wot = (__bf16*)carve((size_t)H_DIM * H_DIM * 2);   // 8 MB
  float*  qf  = (float*)carve((size_t)BATCH * NSEQ * QDIM * 4);   // 4 MB
  float*  kf  = (float*)carve((size_t)BATCH * NSEQ * KDIM * 4);   // 1 MB
  float*  vf  = (float*)carve((size_t)BATCH * NSEQ * KDIM * 4);
  __bf16* qrb = (__bf16*)carve((size_t)NGRP * BATCH * NSEQ * CGRP * 2);  // 32 MB
  __bf16* krb = (__bf16*)carve((size_t)BATCH * NSEQ * CGRP * 2);         // 8 MB
  __bf16* vrb = (__bf16*)carve((size_t)BATCH * NSEQ * CGRP * 2);
  __bf16* cvb = (__bf16*)carve((size_t)BATCH * NSEQ * H_DIM * 2);        // 32 MB

  // 1) weights -> bf16 transposed [n][k]
  wconv_kernel<<<(H_DIM * QDIM + 255) / 256, 256, 0, stream>>>(Wq, Wqt, H_DIM, QDIM);
  wconv_kernel<<<(H_DIM * KDIM + 255) / 256, 256, 0, stream>>>(Wk, Wkt, H_DIM, KDIM);
  wconv_kernel<<<(H_DIM * KDIM + 255) / 256, 256, 0, stream>>>(Wv, Wvt, H_DIM, KDIM);
  wconv_kernel<<<(H_DIM * H_DIM + 255) / 256, 256, 0, stream>>>(Wo, Wot, H_DIM, H_DIM);

  const int Mrows = BATCH * NSEQ;  // 8192
  // 2) projections (f32 activations converted in LDS, bf16 WMMA, f32 out)
  gemm_kernel<true><<<dim3(Mrows / 16, 1), 256, 0, stream>>>(inputs,  Wqt, bq, qf, H_DIM, QDIM);
  gemm_kernel<true><<<dim3(Mrows / 16, 1), 256, 0, stream>>>(context, Wkt, bk, kf, H_DIM, KDIM);
  gemm_kernel<true><<<dim3(Mrows / 16, 1), 256, 0, stream>>>(context, Wvt, bv, vf, H_DIM, KDIM);

  // 3) RoPE + repeat + group
  size_t total = (size_t)NGRP * BATCH * NSEQ * CGRP + 2 * (size_t)BATCH * NSEQ * CGRP;
  rope_kernel<<<(unsigned)((total + 255) / 256), 256, 0, stream>>>(qf, kf, vf, qrb, krb, vrb);

  // 4) windowed attention; attn maps -> d_out tail, AV -> cv (bf16)
  float* outp  = (float*)d_out;
  float* attnp = outp + (size_t)BATCH * NSEQ * H_DIM;
  attn_kernel<<<NGRP * NWIN * BATCH * NHEAD, 128, 0, stream>>>(qrb, krb, vrb, attnp, cvb);

  // 5) output projection -> d_out head (f32)
  gemm_kernel<false><<<dim3(Mrows / 16, H_DIM / 128), 256, 0, stream>>>(cvb, Wot, bo, outp, H_DIM, H_DIM);
}
